// InformerStack_e2e_46918222741808
// MI455X (gfx1250) — matrix operand order repaired and checked by reading, not compiled, not run on hardware
//
#include <hip/hip_runtime.h>
#include <hip/hip_bf16.h>
#include <math.h>

// ---------------------------------------------------------------------------
// InformerStack forward on MI455X (gfx1250, wave32, WMMA).
// All heavy math goes through one BF16 WMMA GEMM (fp32 accumulate):
//   - 128x128x32 block tiles, 8 waves, 32x64 C-tile per wave (2x4 WMMA).
//   - interior blocks: GLOBAL_LOAD_ASYNC_TO_LDS_B128 double-buffered pipeline
//     (ASYNCcnt), overlapping DMA with v_wmma issue.
//   - edge blocks: bounds-checked manual staging.
// Epilogue fuses bias/residual/GELU and emits fp32 + bf16.
// ---------------------------------------------------------------------------

typedef __bf16 bf16_t;
typedef __attribute__((ext_vector_type(16))) __bf16 v16bf;
typedef __attribute__((ext_vector_type(8)))  float  v8f;
typedef __attribute__((ext_vector_type(4)))  unsigned int u32x4;
typedef __attribute__((__vector_size__(4 * sizeof(int)))) int v4i;

#define BM 128
#define BN 128
#define BK 32
#define BKP 40   // BK + 8 bf16 pad -> 80B row stride, keeps 16B alignment

#if __has_builtin(__builtin_amdgcn_global_load_async_to_lds_b128) && \
    __has_builtin(__builtin_amdgcn_s_wait_asynccnt)
#define USE_ASYNC_LDS 1
#endif

#ifdef USE_ASYNC_LDS
__device__ __forceinline__ void async_cp_b128(const bf16_t* g, bf16_t* l) {
  __builtin_amdgcn_global_load_async_to_lds_b128(
      (__attribute__((address_space(1))) v4i*)(bf16_t*)g,
      (__attribute__((address_space(3))) v4i*)l, 0, 0);
}
#endif

struct GemmArgs {
  const bf16_t* A;      // [M,K] row-major (per batch)
  const bf16_t* Bt;     // [N,K] row-major ("transposed B")
  const float*  bias;   // [N] or null
  const float*  resid;  // C-shaped fp32 or null
  float*        C;      // fp32 out or null
  bf16_t*       Cb;     // bf16 out or null
  int M, N, K;
  int lda, ldb, ldc;
  int H;                // z = b*H + h
  long long sAb, sAh, sBb, sBh, sCb, sCh;
  float alpha;
  int act;              // 0=none, 1=gelu(exact)
};

__global__ __launch_bounds__(256, 2)
void gemm_bf16_wmma(GemmArgs g) {
  __shared__ bf16_t As[2][BM][BKP];
  __shared__ bf16_t Bs[2][BN][BKP];   // N-major: Bs[buf][n][k]

  const int z  = blockIdx.z;
  const int bz = z / g.H;
  const int hz = z % g.H;
  const bf16_t* A  = g.A  + (long long)bz * g.sAb + (long long)hz * g.sAh;
  const bf16_t* Bt = g.Bt + (long long)bz * g.sBb + (long long)hz * g.sBh;
  const long long coff = (long long)bz * g.sCb + (long long)hz * g.sCh;

  const int m0 = blockIdx.y * BM;
  const int n0 = blockIdx.x * BN;
  const int tid  = threadIdx.x;
  const int lane = tid & 31;
  const int wave = tid >> 5;     // 8 waves
  const int wm   = wave >> 1;    // 0..3 : 32-row slice
  const int wn   = wave & 1;     // 0..1 : 64-col slice
  const int lr   = lane & 15;
  const int kh   = lane >> 4;

  v8f acc[2][4];
  #pragma unroll
  for (int i = 0; i < 2; ++i)
    #pragma unroll
    for (int j = 0; j < 4; ++j)
      #pragma unroll
      for (int e = 0; e < 8; ++e) acc[i][j][e] = 0.0f;

  // per-thread staging coordinates: 16B chunk at (r0, c0) and (r0+64, c0)
  const int r0 = tid >> 2;          // 0..63
  const int c0 = (tid & 3) * 8;     // 0,8,16,24

  // fragment assembly (ISA 7.12.2 16-bit A 16x32 / B 32x16 layouts) + 8 WMMA
  auto compute_tile = [&](int buf) {
    v16bf afrag[2], bfrag[4];
    #pragma unroll
    for (int tm = 0; tm < 2; ++tm) {
      const bf16_t* p = &As[buf][wm * 32 + tm * 16 + lr][kh * 8];
      union { v16bf v; u32x4 q[2]; } u;
      u.q[0] = *(const u32x4*)p;        // K = kh*8 .. +7
      u.q[1] = *(const u32x4*)(p + 16); // K = 16+kh*8 .. +7
      afrag[tm] = u.v;
    }
    #pragma unroll
    for (int tn = 0; tn < 4; ++tn) {
      const bf16_t* p = &Bs[buf][wn * 64 + tn * 16 + lr][kh * 8];
      union { v16bf v; u32x4 q[2]; } u;
      u.q[0] = *(const u32x4*)p;        // K = kh*8 .. +7
      u.q[1] = *(const u32x4*)(p + 16); // K = 16+kh*8 .. +7
      bfrag[tn] = u.v;
    }
    #pragma unroll
    for (int tm = 0; tm < 2; ++tm)
      #pragma unroll
      for (int tn = 0; tn < 4; ++tn)
        acc[tm][tn] = __builtin_amdgcn_wmma_f32_16x16x32_bf16(
            false, afrag[tm], false, bfrag[tn],
            (short)0, acc[tm][tn], false, false);
  };

  const bool interior = (m0 + BM <= g.M) && (n0 + BN <= g.N);
  // all K in this model are multiples of BK (512/2048/288/1536/64/352/128)

  if (interior) {
#ifdef USE_ASYNC_LDS
    // ---------------- async double-buffered pipeline (ASYNCcnt) ------------
    auto stage = [&](int buf, int k0) {
      const bf16_t* Ap = A  + (long long)(m0 + r0) * g.lda + k0 + c0;
      const bf16_t* Bp = Bt + (long long)(n0 + r0) * g.ldb + k0 + c0;
      async_cp_b128(Ap,                          &As[buf][r0][c0]);
      async_cp_b128(Ap + (long long)64 * g.lda,  &As[buf][r0 + 64][c0]);
      async_cp_b128(Bp,                          &Bs[buf][r0][c0]);
      async_cp_b128(Bp + (long long)64 * g.ldb,  &Bs[buf][r0 + 64][c0]);
    };
    stage(0, 0);
    int buf = 0;
    for (int k0 = 0; k0 < g.K; k0 += BK) {
      __builtin_amdgcn_s_wait_asynccnt(0);  // tile t landed (this wave)
      __syncthreads();                      // all waves: tile t ready; buf^1 free
      if (k0 + BK < g.K) {
        __builtin_prefetch(A  + (long long)(m0 + r0) * g.lda + k0 + BK, 0, 1);
        stage(buf ^ 1, k0 + BK);            // DMA next tile under compute
      }
      compute_tile(buf);
      buf ^= 1;
    }
#else
    // ---------------- manual interior staging (no bounds checks) -----------
    for (int k0 = 0; k0 < g.K; k0 += BK) {
      const bf16_t* Ap = A  + (long long)(m0 + r0) * g.lda + k0 + c0;
      const bf16_t* Bp = Bt + (long long)(n0 + r0) * g.ldb + k0 + c0;
      u32x4 a0 = *(const u32x4*)Ap;
      u32x4 a1 = *(const u32x4*)(Ap + (long long)64 * g.lda);
      u32x4 b0 = *(const u32x4*)Bp;
      u32x4 b1 = *(const u32x4*)(Bp + (long long)64 * g.ldb);
      __syncthreads();                      // previous compute done
      *(u32x4*)&As[0][r0][c0]      = a0;
      *(u32x4*)&As[0][r0 + 64][c0] = a1;
      *(u32x4*)&Bs[0][r0][c0]      = b0;
      *(u32x4*)&Bs[0][r0 + 64][c0] = b1;
      __syncthreads();
      compute_tile(0);
    }
#endif
  } else {
    // ---------------- bounds-checked edge path (single buffer) -------------
    for (int k0 = 0; k0 < g.K; k0 += BK) {
      __syncthreads();
      #pragma unroll
      for (int it = 0; it < 2; ++it) {
        const int r  = r0 + it * 64;
        const int gk = k0 + c0;
        {
          const int gm = m0 + r;
          bf16_t tmp[8];
          if (gm < g.M && (gk + 8) <= g.K) {
            *(u32x4*)tmp = *(const u32x4*)(A + (long long)gm * g.lda + gk);
          } else {
            #pragma unroll
            for (int e = 0; e < 8; ++e)
              tmp[e] = (gm < g.M && (gk + e) < g.K)
                         ? A[(long long)gm * g.lda + gk + e] : (bf16_t)0.0f;
          }
          *(u32x4*)&As[0][r][c0] = *(const u32x4*)tmp;
        }
        {
          const int gn = n0 + r;
          bf16_t tmp[8];
          if (gn < g.N && (gk + 8) <= g.K) {
            *(u32x4*)tmp = *(const u32x4*)(Bt + (long long)gn * g.ldb + gk);
          } else {
            #pragma unroll
            for (int e = 0; e < 8; ++e)
              tmp[e] = (gn < g.N && (gk + e) < g.K)
                         ? Bt[(long long)gn * g.ldb + gk + e] : (bf16_t)0.0f;
          }
          *(u32x4*)&Bs[0][r][c0] = *(const u32x4*)tmp;
        }
      }
      __syncthreads();
      compute_tile(0);
    }
  }

  // epilogue: alpha, bias, residual, activation, dual-precision store
  #pragma unroll
  for (int tm = 0; tm < 2; ++tm) {
    #pragma unroll
    for (int tn = 0; tn < 4; ++tn) {
      const int n = n0 + wn * 64 + tn * 16 + lr;
      #pragma unroll
      for (int j = 0; j < 8; ++j) {
        const int m = m0 + wm * 32 + tm * 16 + kh * 8 + j;
        if (m < g.M && n < g.N) {
          float v = acc[tm][tn][j] * g.alpha;
          if (g.bias)  v += g.bias[n];
          const long long off = coff + (long long)m * g.ldc + n;
          if (g.resid) v += g.resid[off];
          if (g.act == 1) v = 0.5f * v * (1.0f + erff(v * 0.70710678118f));
          if (g.C)  g.C[off]  = v;
          if (g.Cb) g.Cb[off] = (bf16_t)v;
        }
      }
    }
  }
}

// ----------------------------- helper kernels ------------------------------

__device__ inline float block_reduce_sum(float v, float* sh) {
  int t = threadIdx.x;
  sh[t] = v; __syncthreads();
  for (int s = blockDim.x >> 1; s > 0; s >>= 1) {
    if (t < s) sh[t] += sh[t + s];
    __syncthreads();
  }
  float r = sh[0]; __syncthreads();
  return r;
}
__device__ inline float block_reduce_max(float v, float* sh) {
  int t = threadIdx.x;
  sh[t] = v; __syncthreads();
  for (int s = blockDim.x >> 1; s > 0; s >>= 1) {
    if (t < s) sh[t] = fmaxf(sh[t], sh[t + s]);
    __syncthreads();
  }
  float r = sh[0]; __syncthreads();
  return r;
}

__global__ void layernorm_k(const float* x, const float* gw, const float* bw,
                            float* y, bf16_t* yb, int D) {
  __shared__ float sh[256];
  const long long row = blockIdx.x;
  const float* xr = x + row * D;
  float s = 0.f, s2 = 0.f;
  for (int i = threadIdx.x; i < D; i += blockDim.x) { float v = xr[i]; s += v; s2 += v * v; }
  s  = block_reduce_sum(s, sh);
  s2 = block_reduce_sum(s2, sh);
  const float mu = s / D;
  const float rs = rsqrtf(s2 / D - mu * mu + 1e-5f);
  for (int i = threadIdx.x; i < D; i += blockDim.x) {
    float v = (xr[i] - mu) * rs * gw[i] + bw[i];
    if (y)  y[row * D + i]  = v;
    if (yb) yb[row * D + i] = (bf16_t)v;
  }
}

__global__ void softmax_k(const float* S, bf16_t* P, int Lq, int Lk,
                          float alpha, int causal) {
  __shared__ float sh[256];
  const long long row = blockIdx.x;
  const int qi = (int)(row % Lq);
  const float* s = S + row * (long long)Lk;
  bf16_t* p = P + row * (long long)Lk;
  const int lim = causal ? (qi + 1) : Lk;
  float mx = -3.4e38f;
  for (int j = threadIdx.x; j < lim; j += blockDim.x) mx = fmaxf(mx, s[j] * alpha);
  mx = block_reduce_max(mx, sh);
  float sum = 0.f;
  for (int j = threadIdx.x; j < lim; j += blockDim.x) sum += __expf(s[j] * alpha - mx);
  sum = block_reduce_sum(sum, sh);
  const float inv = 1.0f / sum;
  for (int j = threadIdx.x; j < Lk; j += blockDim.x) {
    float v = (j < lim) ? __expf(s[j] * alpha - mx) * inv : 0.0f;
    p[j] = (bf16_t)v;
  }
}

__global__ void wt_k(const float* W, bf16_t* Wt, int K, int N) {
  long long idx = (long long)blockIdx.x * blockDim.x + threadIdx.x;
  long long total = (long long)K * N;
  if (idx >= total) return;
  int k = (int)(idx % K);
  long long n = idx / K;
  Wt[idx] = (bf16_t)W[(long long)k * N + n];
}

__global__ void vtrans_k(const bf16_t* V, bf16_t* Vt, int Bn, int L, int H, int Dh) {
  long long idx = (long long)blockIdx.x * blockDim.x + threadIdx.x;
  long long total = (long long)Bn * H * Dh * L;
  if (idx >= total) return;
  int l = (int)(idx % L);
  long long t = idx / L;
  int d = (int)(t % Dh); t /= Dh;
  int h = (int)(t % H);
  int b = (int)(t / H);
  Vt[idx] = V[((long long)(b * L + l) * H + h) * Dh + d];
}

__global__ void im2col3_k(const float* X, bf16_t* A, int Bn, int L, int C) {
  long long idx = (long long)blockIdx.x * blockDim.x + threadIdx.x;
  long long total = (long long)Bn * L * 3 * C;
  if (idx >= total) return;
  int c  = (int)(idx % C);
  long long t = idx / C;
  int kw = (int)(t % 3); t /= 3;
  int l  = (int)(t % L);
  int b  = (int)(t / L);
  int lp = l + kw - 1;
  if (lp < 0)   lp = L - 1;   // circular pad: prepend x[:, -1:]
  if (lp >= L)  lp = 0;       // append x[:, :1]
  A[idx] = (bf16_t)X[((long long)(b * L + lp)) * C + c];
}

__global__ void addpe_k(float* X, bf16_t* Xb, int Bn, int L, int D) {
  long long idx = (long long)blockIdx.x * blockDim.x + threadIdx.x;
  long long total = (long long)Bn * L * D;
  if (idx >= total) return;
  int d = (int)(idx % D);
  int l = (int)((idx / D) % L);
  float f  = expf(-(float)(d & ~1) * 9.210340371976184f / (float)D);
  float pe = (d & 1) ? cosf((float)l * f) : sinf((float)l * f);
  float v = X[idx] + pe;
  X[idx]  = v;
  Xb[idx] = (bf16_t)v;
}

__global__ void bnstats_k(const float* Hc, float* mean, float* rstd, int Mr, int C) {
  __shared__ float sh[256];
  int c = blockIdx.x;
  float s = 0.f, s2 = 0.f;
  for (int r = threadIdx.x; r < Mr; r += blockDim.x) {
    float v = Hc[(long long)r * C + c]; s += v; s2 += v * v;
  }
  s  = block_reduce_sum(s, sh);
  s2 = block_reduce_sum(s2, sh);
  if (threadIdx.x == 0) {
    float mu = s / Mr;
    mean[c] = mu;
    rstd[c] = rsqrtf(s2 / Mr - mu * mu + 1e-5f);
  }
}

__global__ void bnelu_k(float* Hc, const float* mean, const float* rstd,
                        const float* g, const float* b, long long total, int C) {
  long long idx = (long long)blockIdx.x * blockDim.x + threadIdx.x;
  if (idx >= total) return;
  int c = (int)(idx % C);
  float v = (Hc[idx] - mean[c]) * rstd[c] * g[c] + b[c];
  Hc[idx] = v > 0.f ? v : expm1f(v);
}

__global__ void maxpool_k(const float* Hc, float* Y, bf16_t* Yb, int Bn, int L, int C) {
  int Lo = L / 2;
  long long idx = (long long)blockIdx.x * blockDim.x + threadIdx.x;
  long long total = (long long)Bn * Lo * C;
  if (idx >= total) return;
  int c  = (int)(idx % C);
  long long t = idx / C;
  int lo = (int)(t % Lo);
  int b  = (int)(t / Lo);
  float m = -3.4e38f;
  for (int d = 0; d < 3; ++d) {
    int l = 2 * lo - 1 + d;
    if (l >= 0 && l < L) m = fmaxf(m, Hc[((long long)(b * L + l)) * C + c]);
  }
  long long o = ((long long)(b * Lo + lo)) * C + c;
  Y[o] = m; Yb[o] = (bf16_t)m;
}

__global__ void slice_out_k(const float* X, float* out, int Bn, int L, int Co, int OutL) {
  long long idx = (long long)blockIdx.x * blockDim.x + threadIdx.x;
  long long total = (long long)Bn * OutL * Co;
  if (idx >= total) return;
  int c = (int)(idx % Co);
  long long t = idx / Co;
  int l = (int)(t % OutL);
  int b = (int)(t / OutL);
  out[idx] = X[((long long)(b * L + (L - OutL + l))) * Co + c];
}

// ------------------------------ host helpers -------------------------------

static void launch_gemm(hipStream_t st,
    const bf16_t* A, const bf16_t* Bt, const float* bias, const float* resid,
    float* C, bf16_t* Cb, int M, int N, int K,
    int lda, int ldb, int ldc, int nb, int H,
    long long sAb, long long sAh, long long sBb, long long sBh,
    long long sCb, long long sCh, float alpha, int act) {
  GemmArgs g;
  g.A = A; g.Bt = Bt; g.bias = bias; g.resid = resid; g.C = C; g.Cb = Cb;
  g.M = M; g.N = N; g.K = K; g.lda = lda; g.ldb = ldb; g.ldc = ldc; g.H = H;
  g.sAb = sAb; g.sAh = sAh; g.sBb = sBb; g.sBh = sBh; g.sCb = sCb; g.sCh = sCh;
  g.alpha = alpha; g.act = act;
  dim3 grid((N + BN - 1) / BN, (M + BM - 1) / BM, nb * H);
  gemm_bf16_wmma<<<grid, dim3(256), 0, st>>>(g);
}

static void launch_wt(hipStream_t st, const float* W, bf16_t* Wt, int K, int N) {
  long long total = (long long)K * N;
  wt_k<<<dim3((unsigned)((total + 255) / 256)), dim3(256), 0, st>>>(W, Wt, K, N);
}

struct AttnP { const float *bk,*bo,*bq,*bv,*wk,*wo,*wq,*wv; };

// Dense multi-head attention (prob-sparse approximated densely).
static void attention(hipStream_t st, int Bn, int H, int Dh,
                      const float* xq, const bf16_t* xqb, int Lq,
                      const bf16_t* xkvb, int Lk,
                      const AttnP& p, int causal,
                      bf16_t* wt, bf16_t* qb, bf16_t* kb, bf16_t* vb, bf16_t* vt,
                      float* sc, bf16_t* pb, bf16_t* ctxb, float* outp) {
  const int Dm = H * Dh;
  const int Mq = Bn * Lq, Mk = Bn * Lk;
  launch_wt(st, p.wq, wt, Dm, Dm);
  launch_gemm(st, xqb, wt, p.bq, nullptr, nullptr, qb,
              Mq, Dm, Dm, Dm, Dm, Dm, 1, 1, 0,0,0,0,0,0, 1.f, 0);
  launch_wt(st, p.wk, wt, Dm, Dm);
  launch_gemm(st, xkvb, wt, p.bk, nullptr, nullptr, kb,
              Mk, Dm, Dm, Dm, Dm, Dm, 1, 1, 0,0,0,0,0,0, 1.f, 0);
  launch_wt(st, p.wv, wt, Dm, Dm);
  launch_gemm(st, xkvb, wt, p.bv, nullptr, nullptr, vb,
              Mk, Dm, Dm, Dm, Dm, Dm, 1, 1, 0,0,0,0,0,0, 1.f, 0);
  // scores[b,h] = Q[b,:,h*64:] @ K[b,:,h*64:]^T
  launch_gemm(st, qb, kb, nullptr, nullptr, sc, nullptr,
              Lq, Lk, Dh, Dm, Dm, Lk, Bn, H,
              (long long)Lq * Dm, Dh, (long long)Lk * Dm, Dh,
              (long long)H * Lq * Lk, (long long)Lq * Lk, 1.f, 0);
  softmax_k<<<dim3((unsigned)(Bn * H * Lq)), dim3(256), 0, st>>>(
      sc, pb, Lq, Lk, rsqrtf((float)Dh), causal);
  {
    long long total = (long long)Bn * H * Dh * Lk;
    vtrans_k<<<dim3((unsigned)((total + 255) / 256)), dim3(256), 0, st>>>(
        vb, vt, Bn, Lk, H, Dh);
  }
  launch_gemm(st, pb, vt, nullptr, nullptr, nullptr, ctxb,
              Lq, Dh, Lk, Lk, Lk, Dm, Bn, H,
              (long long)H * Lq * Lk, (long long)Lq * Lk,
              (long long)H * Dh * Lk, (long long)Dh * Lk,
              (long long)Lq * Dm, Dh, 1.f, 0);
  launch_wt(st, p.wo, wt, Dm, Dm);
  launch_gemm(st, ctxb, wt, p.bo, xq, outp, nullptr,
              Mq, Dm, Dm, Dm, Dm, Dm, 1, 1, 0,0,0,0,0,0, 1.f, 0);
}

// ------------------------------- entry point -------------------------------

extern "C" void kernel_launch(void* const* d_in, const int* in_sizes, int n_in,
                              void* d_out, int out_size, void* d_ws, size_t ws_size,
                              hipStream_t stream) {
  (void)in_sizes; (void)n_in; (void)out_size; (void)ws_size;
  const int Bn = 16, Dm = 512, H = 8, Dh = 64, Dff = 2048;
  const int Lenc0 = 512, Ldec = 352, Cin = 96, Co = 96, OutL = 96;

  const float* x_enc = (const float*)d_in[0];
  const float* x_dec = (const float*)d_in[1];

  // ---- params: JAX pytree (dicts flattened in sorted-key order) ----
  int pi = 2;
  auto nf = [&]() { return (const float*)d_in[pi++]; };
  auto getAttn = [&]() { AttnP a; a.bk=nf(); a.bo=nf(); a.bq=nf(); a.bv=nf();
                         a.wk=nf(); a.wo=nf(); a.wq=nf(); a.wv=nf(); return a; };
  const float* dec_emb_b = nf(); const float* dec_emb_w = nf();
  struct DecL { const float *c1b,*c1w,*c2b,*c2w; AttnP cross;
                const float *ln1b,*ln1g,*ln2b,*ln2g,*ln3b,*ln3g; AttnP self_; };
  DecL dec[2];
  for (int j = 0; j < 2; ++j) {
    dec[j].c1b = nf(); dec[j].c1w = nf(); dec[j].c2b = nf(); dec[j].c2w = nf();
    dec[j].cross = getAttn();
    dec[j].ln1b = nf(); dec[j].ln1g = nf(); dec[j].ln2b = nf(); dec[j].ln2g = nf();
    dec[j].ln3b = nf(); dec[j].ln3g = nf();
    dec[j].self_ = getAttn();
  }
  const float* dec_nb = nf(); const float* dec_ng = nf();
  struct ConvP { const float *b,*bnb,*bng,*w; };
  ConvP convs[2];
  for (int i = 0; i < 2; ++i) { convs[i].b = nf(); convs[i].bnb = nf();
                                convs[i].bng = nf(); convs[i].w = nf(); }
  const float* enc_emb_b = nf(); const float* enc_emb_w = nf();
  struct EncL { AttnP a; const float *c1b,*c1w,*c2b,*c2w,*ln1b,*ln1g,*ln2b,*ln2g; };
  EncL enc[3];
  for (int i = 0; i < 3; ++i) {
    enc[i].a.bk = nf(); enc[i].a.bo = nf(); enc[i].a.bq = nf(); enc[i].a.bv = nf();
    enc[i].c1b = nf(); enc[i].c1w = nf(); enc[i].c2b = nf(); enc[i].c2w = nf();
    enc[i].ln1b = nf(); enc[i].ln1g = nf(); enc[i].ln2b = nf(); enc[i].ln2g = nf();
    enc[i].a.wk = nf(); enc[i].a.wo = nf(); enc[i].a.wq = nf(); enc[i].a.wv = nf();
  }
  const float* enc_nb = nf(); const float* enc_ng = nf();
  const float* proj_b = nf(); const float* proj_w = nf();

  // ---- workspace arena ----
  char* cur = (char*)d_ws;
  auto alloc = [&](size_t bytes) {
    void* p = (void*)cur; cur += (bytes + 255) & ~(size_t)255; return p;
  };
  const long long MT = (long long)Bn * Lenc0;                 // 8192 rows max
  float*  s_x   = (float*) alloc(MT * Dm * 4);
  bf16_t* s_xb  = (bf16_t*)alloc(MT * Dm * 2);
  float*  s_y   = (float*) alloc(MT * Dm * 4);
  bf16_t* s_yb  = (bf16_t*)alloc(MT * Dm * 2);
  float*  s_t   = (float*) alloc(MT * Dm * 4);
  float*  s_h   = (float*) alloc(MT * Dff * 4);
  bf16_t* s_hb  = (bf16_t*)alloc(MT * Dff * 2);
  bf16_t* s_qb  = (bf16_t*)alloc(MT * Dm * 2);
  bf16_t* s_kb  = (bf16_t*)alloc(MT * Dm * 2);
  bf16_t* s_vb  = (bf16_t*)alloc(MT * Dm * 2);
  bf16_t* s_vt  = (bf16_t*)alloc(MT * Dm * 2);
  float*  s_sc  = (float*) alloc((long long)Bn * H * Lenc0 * Lenc0 * 4);
  bf16_t* s_pb  = (bf16_t*)alloc((long long)Bn * H * Lenc0 * Lenc0 * 2);
  bf16_t* s_col = (bf16_t*)alloc(MT * 3 * Dm * 2);
  bf16_t* s_wtb = (bf16_t*)alloc((long long)Dff * Dm * 2);
  float*  s_enc = (float*) alloc((long long)Bn * 128 * Dm * 4);
  bf16_t* s_encb= (bf16_t*)alloc((long long)Bn * 128 * Dm * 2);
  float*  s_mean= (float*) alloc(Dm * 4);
  float*  s_rstd= (float*) alloc(Dm * 4);
  float*  s_proj= (float*) alloc((long long)Bn * Ldec * Co * 4);

  auto embed = [&](const float* xin, int L, int Ci, const float* cw, const float* cb) {
    long long tc = (long long)Bn * L * 3 * Ci;
    im2col3_k<<<dim3((unsigned)((tc + 255) / 256)), dim3(256), 0, stream>>>(
        xin, s_col, Bn, L, Ci);
    launch_wt(stream, cw, s_wtb, 3 * Ci, Dm);
    launch_gemm(stream, s_col, s_wtb, cb, nullptr, s_x, nullptr,
                Bn * L, Dm, 3 * Ci, 3 * Ci, 3 * Ci, Dm, 1, 1,
                0,0,0,0,0,0, 1.f, 0);
    long long te = (long long)Bn * L * Dm;
    addpe_k<<<dim3((unsigned)((te + 255) / 256)), dim3(256), 0, stream>>>(
        s_x, s_xb, Bn, L, Dm);
  };

  auto ffn_ln = [&](const float* c1w, const float* c1b,
                    const float* c2w, const float* c2b,
                    const float* lng, const float* lnb, int L) {
    const int M = Bn * L;
    launch_wt(stream, c1w, s_wtb, Dm, Dff);
    launch_gemm(stream, s_yb, s_wtb, c1b, nullptr, nullptr, s_hb,
                M, Dff, Dm, Dm, Dm, Dff, 1, 1, 0,0,0,0,0,0, 1.f, 1); // GELU
    launch_wt(stream, c2w, s_wtb, Dff, Dm);
    launch_gemm(stream, s_hb, s_wtb, c2b, s_y, s_t, nullptr,
                M, Dm, Dff, Dff, Dff, Dm, 1, 1, 0,0,0,0,0,0, 1.f, 0);
    layernorm_k<<<dim3((unsigned)M), dim3(256), 0, stream>>>(
        s_t, lng, lnb, s_x, s_xb, Dm);
  };

  // ================================ encoder ================================
  int L = Lenc0;
  embed(x_enc, L, Cin, enc_emb_w, enc_emb_b);
  for (int i = 0; i < 3; ++i) {
    const int M = Bn * L;
    attention(stream, Bn, H, Dh, s_x, s_xb, L, s_xb, L, enc[i].a, 0,
              s_wtb, s_qb, s_kb, s_vb, s_vt, s_sc, s_pb, s_hb, s_t);
    layernorm_k<<<dim3((unsigned)M), dim3(256), 0, stream>>>(
        s_t, enc[i].ln1g, enc[i].ln1b, s_y, s_yb, Dm);
    ffn_ln(enc[i].c1w, enc[i].c1b, enc[i].c2w, enc[i].c2b,
           enc[i].ln2g, enc[i].ln2b, L);
    if (i < 2) { // distilling conv + BN + ELU + maxpool(3, stride 2, pad 1)
      long long tc = (long long)Bn * L * 3 * Dm;
      im2col3_k<<<dim3((unsigned)((tc + 255) / 256)), dim3(256), 0, stream>>>(
          s_x, s_col, Bn, L, Dm);
      launch_wt(stream, convs[i].w, s_wtb, 3 * Dm, Dm);
      launch_gemm(stream, s_col, s_wtb, convs[i].b, nullptr, s_h, nullptr,
                  Bn * L, Dm, 3 * Dm, 3 * Dm, 3 * Dm, Dm, 1, 1,
                  0,0,0,0,0,0, 1.f, 0);
      bnstats_k<<<dim3((unsigned)Dm), dim3(256), 0, stream>>>(
          s_h, s_mean, s_rstd, Bn * L, Dm);
      long long te = (long long)Bn * L * Dm;
      bnelu_k<<<dim3((unsigned)((te + 255) / 256)), dim3(256), 0, stream>>>(
          s_h, s_mean, s_rstd, convs[i].bng, convs[i].bnb, te, Dm);
      long long tp = (long long)Bn * (L / 2) * Dm;
      maxpool_k<<<dim3((unsigned)((tp + 255) / 256)), dim3(256), 0, stream>>>(
          s_h, s_x, s_xb, Bn, L, Dm);
      L /= 2;
    }
  }
  layernorm_k<<<dim3((unsigned)(Bn * L)), dim3(256), 0, stream>>>(
      s_x, enc_ng, enc_nb, s_enc, s_encb, Dm);
  const int Lenc = L;   // 128

  // ================================ decoder ================================
  embed(x_dec, Ldec, Cin, dec_emb_w, dec_emb_b);
  for (int j = 0; j < 2; ++j) {
    const int M = Bn * Ldec;
    attention(stream, Bn, H, Dh, s_x, s_xb, Ldec, s_xb, Ldec, dec[j].self_, 1,
              s_wtb, s_qb, s_kb, s_vb, s_vt, s_sc, s_pb, s_hb, s_t);
    layernorm_k<<<dim3((unsigned)M), dim3(256), 0, stream>>>(
        s_t, dec[j].ln1g, dec[j].ln1b, s_y, s_yb, Dm);
    attention(stream, Bn, H, Dh, s_y, s_yb, Ldec, s_encb, Lenc, dec[j].cross, 0,
              s_wtb, s_qb, s_kb, s_vb, s_vt, s_sc, s_pb, s_hb, s_t);
    layernorm_k<<<dim3((unsigned)M), dim3(256), 0, stream>>>(
        s_t, dec[j].ln2g, dec[j].ln2b, s_y, s_yb, Dm);
    ffn_ln(dec[j].c1w, dec[j].c1b, dec[j].c2w, dec[j].c2b,
           dec[j].ln3g, dec[j].ln3b, Ldec);
  }
  layernorm_k<<<dim3((unsigned)(Bn * Ldec)), dim3(256), 0, stream>>>(
      s_x, dec_ng, dec_nb, s_y, s_yb, Dm);
  launch_wt(stream, proj_w, s_wtb, Dm, Co);
  launch_gemm(stream, s_yb, s_wtb, proj_b, nullptr, s_proj, nullptr,
              Bn * Ldec, Co, Dm, Dm, Dm, Co, 1, 1, 0,0,0,0,0,0, 1.f, 0);
  long long to = (long long)Bn * OutL * Co;
  slice_out_k<<<dim3((unsigned)((to + 255) / 256)), dim3(256), 0, stream>>>(
      s_proj, (float*)d_out, Bn, Ldec, Co, OutL);
}
